// LSTMClassifier_37099927503212
// MI455X (gfx1250) — compile-verified, matrix operand order checked by reading, not run on hardware
//
#include <hip/hip_runtime.h>
#include <hip/hip_bf16.h>

typedef __bf16 bf16_t;
typedef __attribute__((ext_vector_type(16))) __bf16 v16bf;
typedef __attribute__((ext_vector_type(8)))  float  v8f;

#define B_SZ 64
#define T_SZ 256
#define D_SZ 512
#define H_SZ 1024
#define C_SZ 1000
#define G4H  (4 * H_SZ)

// ---------- helpers ----------

__device__ __forceinline__ bf16_t f2bf(float f) {
    union { float f; unsigned u; } a; a.f = f;
    unsigned u = a.u;
    unsigned r = (u + 0x7FFFu + ((u >> 16) & 1u)) >> 16;   // round-to-nearest-even
    union { unsigned short s; bf16_t b; } o; o.s = (unsigned short)r;
    return o.b;
}

__device__ __forceinline__ float sigmoidf(float x) { return 1.0f / (1.0f + __expf(-x)); }
__device__ __forceinline__ float tanh_fast(float x) {
    float e = __expf(2.0f * x);
    return (e - 1.0f) / (e + 1.0f);
}

union BF16Tile { v16bf v; uint4 q[2]; };

// rowbase already includes (lane>>4)*8 K-half offset; loads K=[k..k+7]+[k+16..k+23]
__device__ __forceinline__ v16bf load_ab(const bf16_t* rowbase, int k) {
    BF16Tile u;
    u.q[0] = *reinterpret_cast<const uint4*>(rowbase + k);
    u.q[1] = *reinterpret_cast<const uint4*>(rowbase + k + 16);
    return u.v;
}

__device__ __forceinline__ v8f wmma_bf16(v16bf a, v16bf b, v8f c) {
    return __builtin_amdgcn_wmma_f32_16x16x32_bf16(
        /*neg_a=*/false, a, /*neg_b=*/false, b,
        /*c_mod=*/(short)0, c, /*reuse_a=*/false, /*reuse_b=*/false);
}

// ---------- fused LSTM timestep ----------
// gates[64, 4H] = Ax[64,Kx] @ Wix[4H,Kx]^T + hprev[64,H] @ Whh[4H,H]^T + bias
// then elementwise cell update for the 16 h-columns this block owns.
// grid = H/16 blocks of 256 threads (8 waves, 2 WMMA tiles each).
__global__ __launch_bounds__(256)
void lstm_step_kernel(const bf16_t* __restrict__ Ax, int Kx,
                      const bf16_t* __restrict__ Wix,
                      const bf16_t* __restrict__ hprev,
                      const bf16_t* __restrict__ Whh,
                      const float*  __restrict__ bias,
                      float*        __restrict__ c,
                      bf16_t*       __restrict__ hnext,
                      bf16_t*       __restrict__ outrow,   // [B,H] or null
                      bf16_t*       __restrict__ hcat)     // pre-offset [B,2H] slice or null
{
    __shared__ float g_lds[4][B_SZ][16];   // 16 KB: i,f,g,o for 64 rows x 16 cols

    const int tid    = threadIdx.x;
    const int wave   = tid >> 5;
    const int lane   = tid & 31;
    const int gate   = wave >> 1;          // 0..3 -> i,f,g,o
    const int mpair  = wave & 1;           // rows 0..31 or 32..63
    const int j0     = blockIdx.x * 16;    // h-column slice
    const int lsel   = lane & 15;
    const int khalf8 = (lane >> 4) * 8;    // ISA 16-bit A/B K-half select

    const int row0 = mpair * 32 + lsel;
    const int row1 = row0 + 16;
    const int nrow = gate * H_SZ + j0 + lsel;  // gate-weight row (= B-matrix column)

    v8f acc0 = {0.f, 0.f, 0.f, 0.f, 0.f, 0.f, 0.f, 0.f};
    v8f acc1 = acc0;

    // phase 1: input projection (K = Kx)
    {
        const bf16_t* a0 = Ax  + (size_t)row0 * Kx + khalf8;
        const bf16_t* a1 = Ax  + (size_t)row1 * Kx + khalf8;
        const bf16_t* bw = Wix + (size_t)nrow * Kx + khalf8;
        for (int k = 0; k < Kx; k += 32) {
            v16bf bv = load_ab(bw, k);
            acc0 = wmma_bf16(load_ab(a0, k), bv, acc0);
            acc1 = wmma_bf16(load_ab(a1, k), bv, acc1);
        }
    }
    // phase 2: recurrent projection (K = H)
    {
        const bf16_t* a0 = hprev + (size_t)row0 * H_SZ + khalf8;
        const bf16_t* a1 = hprev + (size_t)row1 * H_SZ + khalf8;
        const bf16_t* bw = Whh   + (size_t)nrow * H_SZ + khalf8;
        for (int k = 0; k < H_SZ; k += 32) {
            if (k + 64 <= H_SZ) __builtin_prefetch(bw + k + 32, 0, 1);  // global_prefetch_b8
            v16bf bv = load_ab(bw, k);
            acc0 = wmma_bf16(load_ab(a0, k), bv, acc0);
            acc1 = wmma_bf16(load_ab(a1, k), bv, acc1);
        }
    }

    // scatter f32 gate tiles to LDS (ISA C/D layout: M = (lane>>4)*8 + v, N = lane&15)
    #pragma unroll
    for (int v = 0; v < 8; ++v) {
        g_lds[gate][mpair * 32 + khalf8 + v][lsel]      = acc0[v];
        g_lds[gate][mpair * 32 + 16 + khalf8 + v][lsel] = acc1[v];
    }
    __syncthreads();

    // fused LSTM cell: 1024 elements, 4 per thread
    #pragma unroll
    for (int p = 0; p < 4; ++p) {
        int idx = tid + p * 256;
        int b   = idx >> 4;
        int col = idx & 15;
        int n   = j0 + col;
        float ig = g_lds[0][b][col] + bias[n];
        float fg = g_lds[1][b][col] + bias[H_SZ + n];
        float gg = g_lds[2][b][col] + bias[2 * H_SZ + n];
        float og = g_lds[3][b][col] + bias[3 * H_SZ + n];
        float i  = sigmoidf(ig);
        float f  = sigmoidf(fg);
        float gt = tanh_fast(gg);
        float o  = sigmoidf(og);
        size_t off = (size_t)b * H_SZ + n;
        float cN = f * c[off] + i * gt;
        float h  = o * tanh_fast(cN);
        c[off] = cN;
        bf16_t hb = f2bf(h);
        hnext[off] = hb;
        if (outrow) outrow[off] = hb;
        if (hcat)   hcat[(size_t)b * (2 * H_SZ) + n] = hb;
    }
}

// ---------- final FC: logits[64,1000] = hcat[64,2048] @ Wfc[1000,2048]^T + b ----------
// grid = 63 blocks (N padded to 1008), 128 threads (4 waves = 4 M-tiles)
__global__ __launch_bounds__(128)
void fc_kernel(const bf16_t* __restrict__ hcat,
               const bf16_t* __restrict__ Wfc,
               const float*  __restrict__ bfc,
               float*        __restrict__ out)
{
    const int tid    = threadIdx.x;
    const int wave   = tid >> 5;
    const int lane   = tid & 31;
    const int lsel   = lane & 15;
    const int khalf8 = (lane >> 4) * 8;
    const int j0     = blockIdx.x * 16;
    const int K      = 2 * H_SZ;

    int n = j0 + lsel;
    if (n > C_SZ - 1) n = C_SZ - 1;           // clamp OOB weight-row reads; store is guarded

    const bf16_t* a  = hcat + (size_t)(wave * 16 + lsel) * K + khalf8;
    const bf16_t* bw = Wfc  + (size_t)n * K + khalf8;

    v8f acc = {0.f, 0.f, 0.f, 0.f, 0.f, 0.f, 0.f, 0.f};
    for (int k = 0; k < K; k += 32)
        acc = wmma_bf16(load_ab(a, k), load_ab(bw, k), acc);

    #pragma unroll
    for (int v = 0; v < 8; ++v) {
        int row = wave * 16 + khalf8 + v;
        int col = j0 + lsel;
        if (col < C_SZ) out[(size_t)row * C_SZ + col] = acc[v] + bfc[col];
    }
}

// ---------- prologue kernels ----------

__global__ void cvt_f32_bf16_kernel(const float* __restrict__ in, bf16_t* __restrict__ out, int n) {
    int i = blockIdx.x * blockDim.x + threadIdx.x;
    if (i < n) out[i] = f2bf(in[i]);
}

// x[B,T,D] f32 -> xb[T,B,D] bf16 (time-major for the recurrence)
__global__ void transpose_x_kernel(const float* __restrict__ x, bf16_t* __restrict__ xb) {
    int i = blockIdx.x * blockDim.x + threadIdx.x;   // flat over T*B*D (output order)
    int d = i % D_SZ;
    int r = i / D_SZ;
    int b = r % B_SZ;
    int t = r / B_SZ;
    xb[i] = f2bf(x[((size_t)b * T_SZ + t) * D_SZ + d]);
}

__global__ void addvec_kernel(const float* a, const float* b, float* o, int n) {
    int i = blockIdx.x * blockDim.x + threadIdx.x;
    if (i < n) o[i] = a[i] + b[i];
}

__global__ void zero_bf16_kernel(bf16_t* p, int n) {
    int i = blockIdx.x * blockDim.x + threadIdx.x;
    if (i < n) p[i] = f2bf(0.0f);
}

__global__ void zero_f32_kernel(float* p, int n) {
    int i = blockIdx.x * blockDim.x + threadIdx.x;
    if (i < n) p[i] = 0.0f;
}

// ---------- host ----------

extern "C" void kernel_launch(void* const* d_in, const int* in_sizes, int n_in,
                              void* d_out, int out_size, void* d_ws, size_t ws_size,
                              hipStream_t stream) {
    (void)in_sizes; (void)n_in; (void)out_size; (void)ws_size;

    const float* x     = (const float*)d_in[0];
    const float* W_ih0 = (const float*)d_in[1];
    const float* W_hh0 = (const float*)d_in[2];
    const float* b_ih0 = (const float*)d_in[3];
    const float* b_hh0 = (const float*)d_in[4];
    const float* W_ih1 = (const float*)d_in[5];
    const float* W_hh1 = (const float*)d_in[6];
    const float* b_ih1 = (const float*)d_in[7];
    const float* b_hh1 = (const float*)d_in[8];
    const float* W_fc  = (const float*)d_in[9];
    const float* b_fc  = (const float*)d_in[10];
    float* out = (float*)d_out;

    char* ws = (char*)d_ws;
    size_t off = 0;
    auto carve = [&](size_t bytes) -> char* {
        char* p = ws + off;
        off = (off + bytes + 255) & ~(size_t)255;
        return p;
    };

    bf16_t* xb    = (bf16_t*)carve((size_t)T_SZ * B_SZ * D_SZ * 2);   // 16 MB
    bf16_t* outs0 = (bf16_t*)carve((size_t)T_SZ * B_SZ * H_SZ * 2);   // 32 MB
    bf16_t* wih0b = (bf16_t*)carve((size_t)G4H * D_SZ * 2);           //  4 MB
    bf16_t* whh0b = (bf16_t*)carve((size_t)G4H * H_SZ * 2);           //  8 MB
    bf16_t* wih1b = (bf16_t*)carve((size_t)G4H * H_SZ * 2);           //  8 MB
    bf16_t* whh1b = (bf16_t*)carve((size_t)G4H * H_SZ * 2);           //  8 MB
    bf16_t* wfcb  = (bf16_t*)carve((size_t)C_SZ * 2 * H_SZ * 2);      //  4 MB
    float*  bias0 = (float*)carve((size_t)G4H * 4);
    float*  bias1 = (float*)carve((size_t)G4H * 4);
    bf16_t* h0a   = (bf16_t*)carve((size_t)B_SZ * H_SZ * 2);
    bf16_t* h0b   = (bf16_t*)carve((size_t)B_SZ * H_SZ * 2);
    bf16_t* h1a   = (bf16_t*)carve((size_t)B_SZ * H_SZ * 2);
    bf16_t* h1b   = (bf16_t*)carve((size_t)B_SZ * H_SZ * 2);
    float*  c0    = (float*)carve((size_t)B_SZ * H_SZ * 4);
    float*  c1    = (float*)carve((size_t)B_SZ * H_SZ * 4);
    bf16_t* hcat  = (bf16_t*)carve((size_t)B_SZ * 2 * H_SZ * 2);

    int n;
    // prologue: bf16 conversions, bias combine, state init
    n = T_SZ * B_SZ * D_SZ;
    transpose_x_kernel<<<n / 256, 256, 0, stream>>>(x, xb);
    n = G4H * D_SZ;
    cvt_f32_bf16_kernel<<<(n + 255) / 256, 256, 0, stream>>>(W_ih0, wih0b, n);
    n = G4H * H_SZ;
    cvt_f32_bf16_kernel<<<(n + 255) / 256, 256, 0, stream>>>(W_hh0, whh0b, n);
    cvt_f32_bf16_kernel<<<(n + 255) / 256, 256, 0, stream>>>(W_ih1, wih1b, n);
    cvt_f32_bf16_kernel<<<(n + 255) / 256, 256, 0, stream>>>(W_hh1, whh1b, n);
    n = C_SZ * 2 * H_SZ;
    cvt_f32_bf16_kernel<<<(n + 255) / 256, 256, 0, stream>>>(W_fc, wfcb, n);
    n = G4H;
    addvec_kernel<<<(n + 255) / 256, 256, 0, stream>>>(b_ih0, b_hh0, bias0, n);
    addvec_kernel<<<(n + 255) / 256, 256, 0, stream>>>(b_ih1, b_hh1, bias1, n);
    n = B_SZ * H_SZ;
    zero_bf16_kernel<<<(n + 255) / 256, 256, 0, stream>>>(h0a, n);
    zero_bf16_kernel<<<(n + 255) / 256, 256, 0, stream>>>(h1a, n);
    zero_f32_kernel<<<(n + 255) / 256, 256, 0, stream>>>(c0, n);
    zero_f32_kernel<<<(n + 255) / 256, 256, 0, stream>>>(c1, n);

    // layer 0 recurrence (h double-buffered; W resident in L2 across steps)
    for (int t = 0; t < T_SZ; ++t) {
        bf16_t* hp = (t & 1) ? h0b : h0a;
        bf16_t* hn = (t & 1) ? h0a : h0b;
        lstm_step_kernel<<<H_SZ / 16, 256, 0, stream>>>(
            xb + (size_t)t * B_SZ * D_SZ, D_SZ, wih0b, hp, whh0b, bias0,
            c0, hn, outs0 + (size_t)t * B_SZ * H_SZ,
            (t == T_SZ - 1) ? hcat : (bf16_t*)nullptr);
    }
    // layer 1 recurrence
    for (int t = 0; t < T_SZ; ++t) {
        bf16_t* hp = (t & 1) ? h1b : h1a;
        bf16_t* hn = (t & 1) ? h1a : h1b;
        lstm_step_kernel<<<H_SZ / 16, 256, 0, stream>>>(
            outs0 + (size_t)t * B_SZ * H_SZ, H_SZ, wih1b, hp, whh1b, bias1,
            c1, hn, (bf16_t*)nullptr,
            (t == T_SZ - 1) ? hcat + H_SZ : (bf16_t*)nullptr);
    }
    // final classifier
    fc_kernel<<<(C_SZ + 15) / 16, 128, 0, stream>>>(hcat, wfcb, b_fc, out);
}
